// D_RNN_72713796321429
// MI455X (gfx1250) — 8-bit matrix operand order repaired and checked by reading, compiled, not run on hardware
//
#include <hip/hip_runtime.h>
#include <hip/hip_bf16.h>
#include <stdint.h>

#define T_TOT   256
#define COND    192
#define PREDL   (T_TOT - COND)
#define H       256
#define B_TOT   4096

#define ROWS_PER_WG 32
#define WAVES       16
#define THREADS     (WAVES * 32)

#define W_LDS_BYTES (WAVES * 4 * 2 * 2048)
#define ABUF_OFF    W_LDS_BYTES
#define ABUF_BYTES  (4 * 2048)
#define XBUF_OFF    (ABUF_OFF + ABUF_BYTES)
#define OACC_OFF    (XBUF_OFF + ROWS_PER_WG * 4)
#define SMEM_BYTES  (OACC_OFF + ROWS_PER_WG * 4)

typedef __attribute__((ext_vector_type(16))) int   v16i;
typedef __attribute__((ext_vector_type(8)))  float v8f;

__device__ __forceinline__ unsigned char f32_to_e4m3(float f) {
    union { float f; uint32_t u; } t; t.f = f;
    uint32_t u = t.u;
    uint32_t s = (u >> 24) & 0x80u;
    int      e = (int)((u >> 23) & 0xffu);
    uint32_t m = u & 0x7fffffu;
    m += 0x0FFFFFu + ((m >> 20) & 1u);
    if (m >> 23) { m = 0; e++; }
    int ee = e - 127 + 7;
    if (e == 0xff)  return (unsigned char)(s | 0x7e);
    if (ee >= 15)   return (unsigned char)(s | 0x7e);
    if (ee <= 0)    return (unsigned char)s;
    return (unsigned char)(s | ((uint32_t)ee << 3) | (m >> 20));
}

__device__ __forceinline__ float e4m3_to_f32(unsigned char b) {
    int s = b >> 7;
    int e = (b >> 3) & 0xf;
    int m = b & 7;
    float v;
    if (e == 0) {
        v = (float)m * 1.953125e-3f;
    } else {
        union { uint32_t u; float f; } t;
        t.u = ((uint32_t)(e + 120) << 23) | ((uint32_t)m << 20);
        v = t.f;
    }
    return s ? -v : v;
}

__device__ __forceinline__ int b_off(int k, int col) {
    int lane = col + (((k & 31) >= 16) ? 16 : 0);
    int vg   = 4 * (k >> 5) + ((k & 15) >> 2);
    int byte = k & 3;
    return lane * 64 + vg * 4 + byte;
}

__device__ __forceinline__ int a_off(int m, int kk) {
    int h64 = kk >> 6;
    int k6  = kk & 63;
    int g32 = k6 >> 5;
    int r   = k6 & 31;
    int rr  = r & 15;
    int hi  = (r >= 16) ? 1 : 0;
    int lane = m + 16 * hi;
    int vg   = 8 * h64 + 4 * g32 + (rr >> 2);
    int byte = rr & 3;
    return lane * 64 + vg * 4 + byte;
}

__device__ __forceinline__ v16i ldv16(const char* p) {
    const int4 a = *(const int4*)(p);
    const int4 b = *(const int4*)(p + 16);
    const int4 c = *(const int4*)(p + 32);
    const int4 d = *(const int4*)(p + 48);
    v16i r;
    r[0]=a.x;  r[1]=a.y;  r[2]=a.z;  r[3]=a.w;
    r[4]=b.x;  r[5]=b.y;  r[6]=b.z;  r[7]=b.w;
    r[8]=c.x;  r[9]=c.y;  r[10]=c.z; r[11]=c.w;
    r[12]=d.x; r[13]=d.y; r[14]=d.z; r[15]=d.w;
    return r;
}

__device__ __forceinline__ float sigf(float x) {
    return __builtin_amdgcn_rcpf(1.0f + __expf(-x));
}
__device__ __forceinline__ float tanh_fast(float x) {
    return 1.0f - 2.0f * __builtin_amdgcn_rcpf(__expf(2.0f * x) + 1.0f);
}

__global__ void __launch_bounds__(THREADS, 1)
lstm_fp8_wmma_kernel(const float* __restrict__ cd,
                     const float* __restrict__ pred,
                     const float* __restrict__ w_ih,
                     const float* __restrict__ w_hh,
                     const float* __restrict__ b_ih,
                     const float* __restrict__ b_hh,
                     const float* __restrict__ w_out,
                     const float* __restrict__ b_out,
                     float* __restrict__ out)
{
    extern __shared__ char smem[];
    char*  wlds = smem;
    char*  abuf = smem + ABUF_OFF;
    float* xbuf = (float*)(smem + XBUF_OFF);
    float* oacc = (float*)(smem + OACC_OFF);

    const int tid  = threadIdx.x;
    const int w    = tid >> 5;
    const int l    = tid & 31;
    const int ln   = l & 15;
    const int lh   = l >> 4;
    const int row0 = blockIdx.x * ROWS_PER_WG;

    for (int i = tid; i < 4 * H * H; i += THREADS) {
        int n  = i >> 8;
        int k  = i & 255;
        int g  = n >> 8;
        int jn = n & 255;
        int wv = jn >> 4;
        int col = jn & 15;
        int kc = k >> 7, kk = k & 127;
        wlds[wv * 16384 + ((g * 2 + kc) * 2048) + b_off(kk, col)] =
            (char)f32_to_e4m3(w_hh[i]);
    }
    *(int4*)(abuf + tid * 16) = make_int4(0, 0, 0, 0);

    float wih[4], bias[4];
#pragma unroll
    for (int g = 0; g < 4; ++g) {
        int n = g * 256 + w * 16 + ln;
        wih[g]  = w_ih[n];
        bias[g] = b_ih[n] + b_hh[n];
    }

    float cst[2][8];
#pragma unroll
    for (int mt = 0; mt < 2; ++mt)
#pragma unroll
        for (int v = 0; v < 8; ++v) cst[mt][v] = 0.0f;

    __syncthreads();

    const int jcol = w * 16 + ln;
    const int jkc  = jcol >> 7, jkk = jcol & 127;

#pragma unroll 1
    for (int t = 0; t < T_TOT; ++t) {
        if (tid < ROWS_PER_WG) {
            int b = row0 + tid;
            xbuf[tid] = (t < COND) ? cd[b * COND + t]
                                   : pred[b * PREDL + (t - COND)];
            if (t + 1 < T_TOT) {
                int t2 = t + 1;
                const float* p = (t2 < COND) ? &cd[b * COND + t2]
                                             : &pred[b * PREDL + (t2 - COND)];
                __builtin_prefetch(p, 0, 1);
            }
        }
        __syncthreads();

        v8f   acc[2][4];
        float xr[2][8];
#pragma unroll
        for (int mt = 0; mt < 2; ++mt) {
            float4 xa = *(const float4*)&xbuf[mt * 16 + lh * 8];
            float4 xb = *(const float4*)&xbuf[mt * 16 + lh * 8 + 4];
            xr[mt][0]=xa.x; xr[mt][1]=xa.y; xr[mt][2]=xa.z; xr[mt][3]=xa.w;
            xr[mt][4]=xb.x; xr[mt][5]=xb.y; xr[mt][6]=xb.z; xr[mt][7]=xb.w;
#pragma unroll
            for (int g = 0; g < 4; ++g)
#pragma unroll
                for (int v = 0; v < 8; ++v)
                    acc[mt][g][v] = fmaf(xr[mt][v], wih[g], bias[g]);
        }

#pragma unroll
        for (int kc = 0; kc < 2; ++kc) {
            v16i A0 = ldv16(abuf + (0 * 2 + kc) * 2048 + l * 64);
            v16i A1 = ldv16(abuf + (1 * 2 + kc) * 2048 + l * 64);
#pragma unroll
            for (int g = 0; g < 4; ++g) {
                v16i Bt = ldv16(wlds + w * 16384 + (g * 2 + kc) * 2048 + l * 64);
                acc[0][g] = __builtin_amdgcn_wmma_f32_16x16x128_fp8_fp8(
                    A0, Bt, (short)0, acc[0][g], false, false);
                acc[1][g] = __builtin_amdgcn_wmma_f32_16x16x128_fp8_fp8(
                    A1, Bt, (short)0, acc[1][g], false, false);
            }
        }

        float hreg[2][8];
#pragma unroll
        for (int mt = 0; mt < 2; ++mt)
#pragma unroll
            for (int v = 0; v < 8; ++v) {
                float ig = sigf(acc[mt][0][v]);
                float fg = sigf(acc[mt][1][v]);
                float gg = tanh_fast(acc[mt][2][v]);
                float og = sigf(acc[mt][3][v]);
                float c  = fg * cst[mt][v] + ig * gg;
                cst[mt][v] = c;
                hreg[mt][v] = og * tanh_fast(c);
            }

        __syncthreads();

#pragma unroll
        for (int mt = 0; mt < 2; ++mt)
#pragma unroll
            for (int v = 0; v < 8; ++v) {
                int m = lh * 8 + v;
                abuf[(mt * 2 + jkc) * 2048 + a_off(m, jkk)] =
                    (char)f32_to_e4m3(hreg[mt][v]);
            }
    }

    __syncthreads();
    if (tid < ROWS_PER_WG) oacc[tid] = 0.0f;
    __syncthreads();
    {
        int r  = tid & 31;
        int ch = tid >> 5;
        int mt = r >> 4, m = r & 15;
        float p = 0.0f;
#pragma unroll
        for (int jj = 0; jj < 16; ++jj) {
            int j  = ch * 16 + jj;
            int kc = j >> 7, kk = j & 127;
            p += e4m3_to_f32((unsigned char)abuf[(mt * 2 + kc) * 2048 + a_off(m, kk)])
               * w_out[j];
        }
        atomicAdd(&oacc[r], p);
    }
    __syncthreads();
    if (tid < ROWS_PER_WG) {
        float z = oacc[tid] + b_out[0];
        out[row0 + tid] = __builtin_amdgcn_rcpf(1.0f + __expf(-z));
    }
}

extern "C" void kernel_launch(void* const* d_in, const int* in_sizes, int n_in,
                              void* d_out, int out_size, void* d_ws, size_t ws_size,
                              hipStream_t stream) {
    (void)in_sizes; (void)n_in; (void)out_size; (void)d_ws; (void)ws_size;
    const float* cd    = (const float*)d_in[0];
    const float* pred  = (const float*)d_in[1];
    const float* w_ih  = (const float*)d_in[2];
    const float* w_hh  = (const float*)d_in[3];
    const float* b_ih  = (const float*)d_in[4];
    const float* b_hh  = (const float*)d_in[5];
    const float* w_out = (const float*)d_in[6];
    const float* b_out = (const float*)d_in[7];
    float* out = (float*)d_out;

    hipFuncSetAttribute(reinterpret_cast<const void*>(lstm_fp8_wmma_kernel),
                        hipFuncAttributeMaxDynamicSharedMemorySize, SMEM_BYTES);
    dim3 grid(B_TOT / ROWS_PER_WG);
    dim3 block(THREADS);
    lstm_fp8_wmma_kernel<<<grid, block, SMEM_BYTES, stream>>>(
        cd, pred, w_ih, w_hh, b_ih, b_hh, w_out, b_out, out);
}
